// Torso_68805376082617
// MI455X (gfx1250) — compile-verified, matrix operand order checked by reading, not run on hardware
//
#include <hip/hip_runtime.h>
#include <math.h>
#include <stdint.h>

typedef __attribute__((ext_vector_type(16))) _Float16 v16h;
typedef __attribute__((ext_vector_type(8)))  float    v8f;

#define BNUM 512
#define NHEAD 16
#define NW 4                          // waves per block in attn_step
#define SCALE 0.17677669529663687f   // 1/sqrt(32)

// K-index of element e of an f16 A/B fragment for v_wmma_*_16x16x32_* (wave32).
// A 16x32: lanes 0-15 hold rows, VGPR0-3 => K 0..7 (lanes<16) / 8..15 (lanes>=16),
// VGPR4-7 => K 16..23 / 24..31; two f16 per VGPR. B (32x16) symmetric, N per lane.
__device__ __forceinline__ int kmap(int e, int lane) {
    int vg = e >> 1;
    return ((vg & 4) ? 16 : 0) + ((lane & 16) ? 8 : 0) + ((vg & 3) << 1) + (e & 1);
}

// Wave-local LDS handoff: wave32 is lockstep, LDS ops per wave are in-order,
// so draining DScnt with a compiler barrier is a full wave-local sync for
// wave-private LDS regions (no block barrier needed).
__device__ __forceinline__ void wave_lds_sync() {
    __builtin_amdgcn_wave_barrier();
    asm volatile("s_wait_dscnt 0" ::: "memory");
    __builtin_amdgcn_wave_barrier();
}

// Async global->LDS copy of 4 bytes (CDNA5 GLOBAL_LOAD_ASYNC_TO_LDS_B32, GV mode).
// Generic pointers to LDS carry the LDS byte offset in their low 32 bits.
__device__ __forceinline__ void async_copy4(float* lds_dst, const float* gsrc) {
    uint32_t l = (uint32_t)(uintptr_t)lds_dst;
    uint64_t g = (uint64_t)(uintptr_t)gsrc;
    asm volatile("global_load_async_to_lds_b32 %0, %1, off"
                 :: "v"(l), "v"(g) : "memory");
}

__device__ __forceinline__ void async_drain() {
    asm volatile("s_wait_asynccnt 0" ::: "memory");
}

// ---------------------------------------------------------------------------
// Stage 1: gi = [transpose(input_t), p] ; x[i] = gi @ g2g_w[i] + g2g_b[i]
// grid (512, 3), block 128
// ---------------------------------------------------------------------------
__global__ void g2g_kernel(const float* __restrict__ input_t,
                           const float* __restrict__ input_s,
                           const float* __restrict__ s2g_w,
                           const float* __restrict__ s2g_b,
                           const float* __restrict__ g2g_w,
                           const float* __restrict__ g2g_b,
                           float* __restrict__ xbuf) {
    __shared__ float gi[464];
    const int b = blockIdx.x;
    const int i = blockIdx.y;
    const int tid = threadIdx.x;
    const float is0 = input_s[b * 3 + 0];
    const float is1 = input_s[b * 3 + 1];
    const float is2 = input_s[b * 3 + 2];
    for (int g = tid; g < 464; g += 128) {
        int a12 = g / 29;
        int q   = g - a12 * 29;
        int a1 = a12 >> 2, a2 = a12 & 3;
        float v;
        if (q == 28) {
            v = is0 * s2g_w[i * 48 +       a12] +
                is1 * s2g_w[i * 48 + 16 +  a12] +
                is2 * s2g_w[i * 48 + 32 +  a12] +
                s2g_b[i * 16 + a12];
        } else {
            int hi = q / 7, t = q - hi * 7;
            int s1, s2, s3;
            if (i == 0)      { s1 = a1; s2 = a2; s3 = hi; }
            else if (i == 1) { s3 = a1; s1 = a2; s2 = hi; }
            else             { s2 = a1; s3 = a2; s1 = hi; }
            v = input_t[(((b * 7 + t) * 4 + s1) * 4 + s2) * 4 + s3];
        }
        gi[g] = v;
    }
    __syncthreads();
    if (tid < 48) {
        float acc = g2g_b[i * 48 + tid];
        const float* wcol = g2g_w + (size_t)i * 464 * 48 + tid;
        #pragma unroll 8
        for (int g = 0; g < 464; ++g) acc += gi[g] * wcol[g * 48];
        xbuf[(i * BNUM + b) * 48 + tid] = acc;   // layout [i][b][a12][c]
    }
}

// ---------------------------------------------------------------------------
// Stage 2: one (n, m1, m2) attention step.
// grid (64, 4), block 128 (4 waves). Wave w handles 2 batches => 16 token rows
// packed into one 16x16 WMMA tile (rows 0-7 batch b0, rows 8-15 batch b1).
// ---------------------------------------------------------------------------
__global__ void __launch_bounds__(128)
attn_step(float* __restrict__ xbuf,
          const float* __restrict__ lnx_w, const float* __restrict__ lnx_b,
          const float* __restrict__ lny_w, const float* __restrict__ lny_b,
          const float* __restrict__ lnf_w, const float* __restrict__ lnf_b,
          const float* __restrict__ wq, const float* __restrict__ bq,
          const float* __restrict__ wk, const float* __restrict__ bk,
          const float* __restrict__ wv, const float* __restrict__ bv,
          const float* __restrict__ w1, const float* __restrict__ b1,
          const float* __restrict__ w2, const float* __restrict__ b2,
          const float* __restrict__ w3, const float* __restrict__ b3,
          int n, int m1, int m2) {
    __shared__ float s_wq[1536], s_wk[1536], s_wv[1536], s_w1[1536];
    __shared__ float s_bq[512], s_bk[512], s_bv[512];
    __shared__ float s_w2[36], s_w3[36], s_b2[12], s_b1[3], s_b3[3];
    __shared__ float s_lnxw[3], s_lnxb[3], s_lnyw[3], s_lnyb[3], s_lnfw[3], s_lnfb[3];
    __shared__ float s_x[NW][16][3], s_xn[NW][16][3], s_yn[NW][16][3];
    __shared__ _Float16 s_sc[NW][256];   // per-wave 16x16 score tile (f16)
    __shared__ float s_oh[NW][512];      // per-wave 16x32 per-head O tile

    const int tid  = threadIdx.x;
    const int w    = tid >> 5;
    const int lane = tid & 31;
    const int s    = blockIdx.y;
    const int ns   = n * 4 + s;

    // ---- async-stage the big weights for this (n,s) into LDS (ASYNCcnt) ----
    {
        const float* gwq = wq + ns * 1536;
        const float* gwk = wk + ns * 1536;
        const float* gwv = wv + ns * 1536;
        const float* gw1 = w1 + ns * 1536;
        for (int i = tid; i < 1536; i += 128) {
            async_copy4(&s_wq[i], gwq + i);
            async_copy4(&s_wk[i], gwk + i);
            async_copy4(&s_wv[i], gwv + i);
            async_copy4(&s_w1[i], gw1 + i);
        }
        const float* gbq = bq + ns * 512;
        const float* gbk = bk + ns * 512;
        const float* gbv = bv + ns * 512;
        for (int i = tid; i < 512; i += 128) {
            async_copy4(&s_bq[i], gbq + i);
            async_copy4(&s_bk[i], gbk + i);
            async_copy4(&s_bv[i], gbv + i);
        }
    }
    // small params: plain loads
    if (tid < 36) { s_w2[tid] = w2[ns * 36 + tid]; s_w3[tid] = w3[ns * 36 + tid]; }
    if (tid < 12) s_b2[tid] = b2[ns * 12 + tid];
    if (tid < 3) {
        s_b1[tid]   = b1[ns * 3 + tid];   s_b3[tid]   = b3[ns * 3 + tid];
        s_lnxw[tid] = lnx_w[ns * 3 + tid]; s_lnxb[tid] = lnx_b[ns * 3 + tid];
        s_lnyw[tid] = lny_w[ns * 3 + tid]; s_lnyb[tid] = lny_b[ns * 3 + tid];
        s_lnfw[tid] = lnf_w[ns * 3 + tid]; s_lnfb[tid] = lnf_b[ns * 3 + tid];
    }

    // ---- stage the 16 token rows: a[b,s,j] = j<4 ? x[m1][b,s,j] : x[m2][b,j-4,s]
    const int b0 = blockIdx.x * (2 * NW) + w * 2;
    for (int idx = lane; idx < 48; idx += 32) {
        int r = idx / 3, c = idx - r * 3;
        int bb = r >> 3, t = r & 7, b = b0 + bb;
        float v;
        if (t < 4) v = xbuf[((m1 * BNUM + b) * 16 + s * 4 + t) * 3 + c];
        else       v = xbuf[((m2 * BNUM + b) * 16 + (t - 4) * 4 + s) * 3 + c];
        s_x[w][r][c] = v;
    }
    async_drain();      // this wave's async weight tiles are in LDS
    __syncthreads();    // all waves' weight contributions visible block-wide

    // ---- LayerNorm (C=3) into xn / yn (wave-private) ----
    if (lane < 16) {
        int r = lane;
        float x0 = s_x[w][r][0], x1 = s_x[w][r][1], x2 = s_x[w][r][2];
        float mu = (x0 + x1 + x2) * (1.0f / 3.0f);
        float d0 = x0 - mu, d1 = x1 - mu, d2 = x2 - mu;
        float inv = rsqrtf((d0 * d0 + d1 * d1 + d2 * d2) * (1.0f / 3.0f) + 1e-5f);
        s_xn[w][r][0] = d0 * inv * s_lnxw[0] + s_lnxb[0];
        s_xn[w][r][1] = d1 * inv * s_lnxw[1] + s_lnxb[1];
        s_xn[w][r][2] = d2 * inv * s_lnxw[2] + s_lnxb[2];
        s_yn[w][r][0] = d0 * inv * s_lnyw[0] + s_lnyb[0];
        s_yn[w][r][1] = d1 * inv * s_lnyw[1] + s_lnyb[1];
        s_yn[w][r][2] = d2 * inv * s_lnyw[2] + s_lnyb[2];
    }
    wave_lds_sync();

    // out-projection ownership (stable across heads): idx = token*3 + c
    const int tok0 = lane / 3, c0 = lane % 3;           // lane < 32 < 48 always
    const int idx1 = lane + 32;
    const bool v1  = idx1 < 48;
    const int tok1 = idx1 / 3, c1 = idx1 % 3;
    float acc0 = 0.0f, acc1 = 0.0f;

    const int m     = lane & 15;               // token row (A) / token col (B)
    const int half8 = (lane & 16) ? 8 : 0;     // C/D layout M offset
    const int ncol  = lane & 15;

    for (int h = 0; h < NHEAD; ++h) {
        const int base = h * 32;
        const float xm0 = s_xn[w][m][0], xm1 = s_xn[w][m][1], xm2 = s_xn[w][m][2];
        const float ym0 = s_yn[w][m][0], ym1 = s_yn[w][m][1], ym2 = s_yn[w][m][2];

        // A = q (16 tok x 32 d), B = k^T (32 d x 16 tok); projections inline
        v16h fa, fb;
        #pragma unroll
        for (int e = 0; e < 16; ++e) {
            int K = kmap(e, lane);
            int hd = base + K;
            fa[e] = (_Float16)(xm0 * s_wq[hd] + xm1 * s_wq[512 + hd] +
                               xm2 * s_wq[1024 + hd] + s_bq[hd]);
            fb[e] = (_Float16)(ym0 * s_wk[hd] + ym1 * s_wk[512 + hd] +
                               ym2 * s_wk[1024 + hd] + s_bk[hd]);
        }
        v8f csc = {};
        csc = __builtin_amdgcn_wmma_f32_16x16x32_f16(false, fa, false, fb,
                                                     (short)0, csc, false, false);
        // keep only the per-batch diagonal 8x8 blocks, apply 1/sqrt(d)
        const float keep = ((lane & 31) < 8 || (lane & 31) >= 24) ? SCALE : 0.0f;
        #pragma unroll
        for (int r = 0; r < 8; ++r)
            s_sc[w][(half8 + r) * 16 + ncol] = (_Float16)(csc[r] * keep);
        wave_lds_sync();

        // A = scores (16x16, K padded to 32 with zeros)
        v16h fs;
        #pragma unroll
        for (int e = 0; e < 16; ++e) {
            int K = kmap(e, lane);
            fs[e] = (K < 16) ? s_sc[w][m * 16 + K] : (_Float16)0.0f;
        }
        // o = scores @ v, two 16-wide d halves
        #pragma unroll
        for (int half = 0; half < 2; ++half) {
            const int d = base + half * 16 + ncol;
            v16h fv;
            #pragma unroll
            for (int e = 0; e < 16; ++e) {
                int K = kmap(e, lane);
                if (K < 16) {
                    float y0 = s_yn[w][K][0], y1 = s_yn[w][K][1], y2 = s_yn[w][K][2];
                    fv[e] = (_Float16)(y0 * s_wv[d] + y1 * s_wv[512 + d] +
                                       y2 * s_wv[1024 + d] + s_bv[d]);
                } else {
                    fv[e] = (_Float16)0.0f;
                }
            }
            v8f co = {};
            co = __builtin_amdgcn_wmma_f32_16x16x32_f16(false, fs, false, fv,
                                                        (short)0, co, false, false);
            #pragma unroll
            for (int r = 0; r < 8; ++r)
                s_oh[w][(half8 + r) * 32 + half * 16 + ncol] = co[r];
        }
        wave_lds_sync();

        // out-projection partial: acc += o_head @ w1[base:base+32]
        {
            float a = 0.0f;
            #pragma unroll 8
            for (int d = 0; d < 32; ++d)
                a += s_oh[w][tok0 * 32 + d] * s_w1[(base + d) * 3 + c0];
            acc0 += a;
            if (v1) {
                float bacc = 0.0f;
                #pragma unroll 8
                for (int d = 0; d < 32; ++d)
                    bacc += s_oh[w][tok1 * 32 + d] * s_w1[(base + d) * 3 + c1];
                acc1 += bacc;
            }
        }
        wave_lds_sync();
    }

    // residual add + b1 (wave-private)
    s_x[w][tok0][c0] += acc0 + s_b1[c0];
    if (v1) s_x[w][tok1][c1] += acc1 + s_b1[c1];
    wave_lds_sync();

    // FFN: x + gelu(ln(x)@w2 + b2)@w3 + b3 ; write back to global state
    if (lane < 16) {
        int r = lane;
        float x0 = s_x[w][r][0], x1 = s_x[w][r][1], x2 = s_x[w][r][2];
        float mu = (x0 + x1 + x2) * (1.0f / 3.0f);
        float d0 = x0 - mu, d1 = x1 - mu, d2 = x2 - mu;
        float inv = rsqrtf((d0 * d0 + d1 * d1 + d2 * d2) * (1.0f / 3.0f) + 1e-5f);
        float f0 = d0 * inv * s_lnfw[0] + s_lnfb[0];
        float f1 = d1 * inv * s_lnfw[1] + s_lnfb[1];
        float f2 = d2 * inv * s_lnfw[2] + s_lnfb[2];
        float o0 = 0.0f, o1 = 0.0f, o2 = 0.0f;
        #pragma unroll
        for (int j = 0; j < 12; ++j) {
            float t = f0 * s_w2[j] + f1 * s_w2[12 + j] + f2 * s_w2[24 + j] + s_b2[j];
            float g = 0.5f * t * (1.0f + erff(t * 0.70710678118654752f));
            o0 += g * s_w3[j * 3 + 0];
            o1 += g * s_w3[j * 3 + 1];
            o2 += g * s_w3[j * 3 + 2];
        }
        float r0 = x0 + o0 + s_b3[0];
        float r1 = x1 + o1 + s_b3[1];
        float r2 = x2 + o2 + s_b3[2];
        int bb = r >> 3, t = r & 7, b = b0 + bb;
        float* dst;
        if (t < 4) dst = xbuf + ((m1 * BNUM + b) * 16 + s * 4 + t) * 3;
        else       dst = xbuf + ((m2 * BNUM + b) * 16 + (t - 4) * 4 + s) * 3;
        dst[0] = r0; dst[1] = r1; dst[2] = r2;
    }
}

// ---------------------------------------------------------------------------
// Stage 3: pack state [i][b][16][3] -> out (B, 48, 3)
// ---------------------------------------------------------------------------
__global__ void pack_kernel(const float* __restrict__ xbuf,
                            float* __restrict__ out, int n) {
    int idx = blockIdx.x * blockDim.x + threadIdx.x;
    if (idx >= n) return;
    int c = idx % 3;
    int rest = idx / 3;
    int g48 = rest % 48;
    int b = rest / 48;
    int i = g48 / 16, a12 = g48 % 16;
    out[idx] = xbuf[(i * BNUM + b) * 48 + a12 * 3 + c];
}

extern "C" void kernel_launch(void* const* d_in, const int* in_sizes, int n_in,
                              void* d_out, int out_size, void* d_ws, size_t ws_size,
                              hipStream_t stream) {
    const float* input_t = (const float*)d_in[0];
    const float* input_s = (const float*)d_in[1];
    const float* s2g_w   = (const float*)d_in[2];
    const float* s2g_b   = (const float*)d_in[3];
    const float* g2g_w   = (const float*)d_in[4];
    const float* g2g_b   = (const float*)d_in[5];
    const float* lnx_w   = (const float*)d_in[6];
    const float* lnx_b   = (const float*)d_in[7];
    const float* lny_w   = (const float*)d_in[8];
    const float* lny_b   = (const float*)d_in[9];
    const float* lnf_w   = (const float*)d_in[10];
    const float* lnf_b   = (const float*)d_in[11];
    const float* wq = (const float*)d_in[12];
    const float* bq = (const float*)d_in[13];
    const float* wk = (const float*)d_in[14];
    const float* bk = (const float*)d_in[15];
    const float* wv = (const float*)d_in[16];
    const float* bv = (const float*)d_in[17];
    const float* w1 = (const float*)d_in[18];
    const float* b1 = (const float*)d_in[19];
    const float* w2 = (const float*)d_in[20];
    const float* b2 = (const float*)d_in[21];
    const float* w3 = (const float*)d_in[22];
    const float* b3 = (const float*)d_in[23];

    float* xbuf = (float*)d_ws;   // 3*512*16*3 floats = 294912 bytes of state

    g2g_kernel<<<dim3(512, 3), 128, 0, stream>>>(input_t, input_s, s2g_w, s2g_b,
                                                 g2g_w, g2g_b, xbuf);

    static const int pairs[3][2] = {{0, 1}, {2, 0}, {1, 2}};
    for (int n = 0; n < 8; ++n) {
        for (int p = 0; p < 3; ++p) {
            attn_step<<<dim3(64, 4), 128, 0, stream>>>(
                xbuf, lnx_w, lnx_b, lny_w, lny_b, lnf_w, lnf_b,
                wq, bq, wk, bk, wv, bv, w1, b1, w2, b2, w3, b3,
                n, pairs[p][0], pairs[p][1]);
        }
    }

    int total = BNUM * 48 * 3;
    pack_kernel<<<(total + 255) / 256, 256, 0, stream>>>(xbuf, (float*)d_out, total);
}
